// LSQAttention_57483842289861
// MI455X (gfx1250) — matrix operand order repaired and checked by reading, not compiled, not run on hardware
//
#include <hip/hip_runtime.h>
#include <hip/hip_bf16.h>

typedef __attribute__((ext_vector_type(16))) __bf16 v16bf;
typedef __attribute__((ext_vector_type(8)))  float  v8f;

#define EPSV 1e-5f

union Frag {            // one WMMA 16x32 bf16 operand: two 128-bit halves
  v16bf v;
  uint4 q[2];
};
union Pack8 {           // 8 bf16 -> one 128-bit store
  uint4  q;
  __bf16 e[8];
};

__device__ __forceinline__ uint4 ld128(const __bf16* p) {
  return *reinterpret_cast<const uint4*>(p);
}
__device__ __forceinline__ v8f wmma_bf16(v16bf a, v16bf b, v8f c) {
  return __builtin_amdgcn_wmma_f32_16x16x32_bf16(false, a, false, b, (short)0, c,
                                                 false, false);
}
// issue one per-lane 16B async global->LDS copy (ASYNCcnt++)
__device__ __forceinline__ void async_g2l_b128(const __bf16* gp, const __bf16* lp) {
  unsigned loff = (unsigned)(size_t)(const void*)lp;            // LDS byte offset
  unsigned long long ga = (unsigned long long)(size_t)(const void*)gp;
  asm volatile("global_load_async_to_lds_b128 %0, %1, off"
               :: "v"(loff), "v"(ga) : "memory");
}

// ---------------------------------------------------------------------------
// Weight f32 -> bf16 (contiguous)
// ---------------------------------------------------------------------------
__global__ __launch_bounds__(256)
void cvtw_kernel(const float* __restrict__ W, __bf16* __restrict__ Wb, int n) {
  int i = blockIdx.x * 256 + threadIdx.x;
  if (i < n) Wb[i] = (__bf16)W[i];
}

// ---------------------------------------------------------------------------
// x (B,256,1024) f32  ->  XT (B,1024,256) bf16, LDS-tiled 16x16 transpose
// ---------------------------------------------------------------------------
__global__ __launch_bounds__(256)
void xpose_kernel(const float* __restrict__ X, __bf16* __restrict__ XT) {
  __shared__ float t[16][17];
  const int n0 = blockIdx.x * 16, c0 = blockIdx.y * 16, b = blockIdx.z;
  const int tn = threadIdx.x & 15, tc = threadIdx.x >> 4;
  t[tc][tn] = X[((size_t)b * 256 + c0 + tc) * 1024 + n0 + tn];
  __syncthreads();
  const int wc = threadIdx.x & 15, wn = threadIdx.x >> 4;
  XT[((size_t)b * 1024 + n0 + wn) * 256 + c0 + wc] = (__bf16)t[wc][wn];
}

// ---------------------------------------------------------------------------
// GEMM + BN:  acc[o,n] = sum_c W[o,c] * XT[n,c]   (bf16 in, f32 acc)
// ---------------------------------------------------------------------------
__global__ __launch_bounds__(32)
void gemm16_bn_kernel(const __bf16* __restrict__ XT, const __bf16* __restrict__ W,
                      const float* __restrict__ gg, const float* __restrict__ bb,
                      const float* __restrict__ mm, const float* __restrict__ vv,
                      float*  __restrict__ outF,   // natural (B,COUT,1024) f32
                      __bf16* __restrict__ outN,   // natural (B,COUT,1024) bf16
                      __bf16* __restrict__ outT,   // transposed (B,1024,COUT) bf16
                      int CIN, int COUT) {
  const int lane = threadIdx.x;
  const int li = lane & 15;
  const int hi = lane >> 4;
  const int n0 = blockIdx.x * 16;
  const int o0 = blockIdx.y * 16;
  const int b  = blockIdx.z;

  const __bf16* wrow = W + (size_t)(o0 + li) * CIN;
  const __bf16* xrow = XT + ((size_t)b * 1024 + n0 + li) * CIN;

  v8f acc = {0.f, 0.f, 0.f, 0.f, 0.f, 0.f, 0.f, 0.f};
  for (int k0 = 0; k0 < CIN; k0 += 32) {
    Frag a, bt;
    a.q[0]  = ld128(wrow + k0 + hi * 8);
    a.q[1]  = ld128(wrow + k0 + 16 + hi * 8);
    bt.q[0] = ld128(xrow + k0 + hi * 8);
    bt.q[1] = ld128(xrow + k0 + 16 + hi * 8);
    acc = wmma_bf16(a.v, bt.v, acc);
  }

  float val[8];
#pragma unroll
  for (int r = 0; r < 8; ++r) {
    const int o = o0 + r + hi * 8;
    const float s = gg[o] * rsqrtf(vv[o] + EPSV);
    const float t = bb[o] - mm[o] * s;
    val[r] = acc[r] * s + t;
  }
  if (outF) {
#pragma unroll
    for (int r = 0; r < 8; ++r)
      outF[((size_t)b * COUT + o0 + r + hi * 8) * 1024 + n0 + li] = val[r];
  }
  if (outN) {
#pragma unroll
    for (int r = 0; r < 8; ++r)
      outN[((size_t)b * COUT + o0 + r + hi * 8) * 1024 + n0 + li] = (__bf16)val[r];
  }
  if (outT) {
    Pack8 p;
#pragma unroll
    for (int r = 0; r < 8; ++r) p.e[r] = (__bf16)val[r];
    *reinterpret_cast<uint4*>(outT + ((size_t)b * 1024 + n0 + li) * COUT +
                              o0 + hi * 8) = p.q;
  }
}

// ---------------------------------------------------------------------------
// Cooperative flash attention. Workgroup = 256 threads per (b, h, quarter).
// K/V staged through LDS in 4 double-buffered m-chunks of 256 keys via
// global_load_async_to_lds_b128 (ASYNCcnt pipelined with s_wait_asynccnt).
// Each wave owns 2 query tiles and shares the K/V fragments between them.
// ---------------------------------------------------------------------------
__global__ __launch_bounds__(256)
void attn_kernel(const __bf16* __restrict__ qkvT, const __bf16* __restrict__ qkvN,
                 __bf16* __restrict__ yT) {
  __shared__ alignas(16) __bf16 kch[2][256 * 16];  // [m][c]  8 KB per buffer
  __shared__ alignas(16) __bf16 vch[2][32 * 256];  // [cv][m] 16 KB per buffer

  const int tid  = threadIdx.x;
  const int lane = tid & 31;
  const int w    = tid >> 5;
  const int li = lane & 15, hi = lane >> 4;
  const int h = blockIdx.y, b = blockIdx.z;
  const int tile0 = blockIdx.x * 16 + w * 2;       // first of 2 query tiles

  const __bf16* kg = qkvT + (size_t)b * 1024 * 512 + h * 64 + 16;
  const __bf16* vg = qkvN + ((size_t)b * 512 + h * 64 + 32) * 1024;
  const uint4 z4 = {0u, 0u, 0u, 0u};
  const v8f cz = {0.f, 0.f, 0.f, 0.f, 0.f, 0.f, 0.f, 0.f};

  // stage one 256-key chunk: 6 async wave-ops (2 for K, 4 for V)
  auto stage = [&](int ci, int buf) {
    const int M0 = ci * 256;
#pragma unroll
    for (int j = 0; j < 2; ++j) {                  // K: 512 x 16B chunks
      const int idx = tid + j * 256;
      const int m = idx >> 1, half = idx & 1;
      async_g2l_b128(kg + (size_t)(M0 + m) * 512 + half * 8,
                     &kch[buf][m * 16 + half * 8]);
    }
#pragma unroll
    for (int j = 0; j < 4; ++j) {                  // V: 1024 x 16B chunks
      const int idx = tid + j * 256;
      const int cv = idx >> 5, mo = (idx & 31) * 8;
      async_g2l_b128(vg + (size_t)cv * 1024 + M0 + mo,
                     &vch[buf][cv * 256 + mo]);
    }
  };

  // B operands: q for the two query tiles (upper 8 elements are K-padding -> zero)
  Frag bq0, bq1;
  {
    const __bf16* q0 = qkvT + ((size_t)b * 1024 + tile0 * 16 + li) * 512 + h * 64;
    const __bf16* q1 = q0 + 16 * 512;
    uint4 a0 = ld128(q0 + hi * 8);
    uint4 b0 = ld128(q1 + hi * 8);
    bq0.q[0] = a0;  bq0.q[1] = z4;
    bq1.q[0] = b0;  bq1.q[1] = z4;
  }

  v8f O00 = cz, O01 = cz, O10 = cz, O11 = cz;
  float mr0 = -1e30f, lr0 = 0.f, mr1 = -1e30f, lr1 = 0.f;
  const float scale = 0.25f;  // KD^-0.5

  // one online-softmax step for one query tile
  auto step = [&](const v16bf& bqv, const v16bf& ak0v, const v16bf& ak1v,
                  const v16bf& av0v, const v16bf& av1v,
                  v8f& Oa, v8f& Ob, float& mrun, float& lrun) {
    v8f s0 = wmma_bf16(ak0v, bqv, cz);
    v8f s1 = wmma_bf16(ak1v, bqv, cz);
    float lm = -1e30f;
#pragma unroll
    for (int r = 0; r < 8; ++r) {
      s0[r] *= scale; s1[r] *= scale;
      lm = fmaxf(lm, fmaxf(s0[r], s1[r]));
    }
    lm = fmaxf(lm, __shfl_xor(lm, 16, 32));
    const float mnew = fmaxf(mrun, lm);
    const float fsc = __expf(mrun - mnew);
    float ls = 0.f;
#pragma unroll
    for (int r = 0; r < 8; ++r) {
      s0[r] = __expf(s0[r] - mnew);
      s1[r] = __expf(s1[r] - mnew);
      ls += s0[r] + s1[r];
    }
    ls += __shfl_xor(ls, 16, 32);
    lrun = lrun * fsc + ls;
    mrun = mnew;
    v16bf bp;
#pragma unroll
    for (int r = 0; r < 8; ++r) {
      const float a0 = s0[r], a1 = s1[r];
      const float p0 = __shfl_xor(a0, 16, 32);
      const float p1 = __shfl_xor(a1, 16, 32);
      bp[r]     = (__bf16)a0;
      bp[r + 8] = (__bf16)a1;
    }
#pragma unroll
    for (int r = 0; r < 8; ++r) { Oa[r] *= fsc; Ob[r] *= fsc; }
    Oa = wmma_bf16(av0v, bp, Oa);
    Ob = wmma_bf16(av1v, bp, Ob);
  };

  stage(0, 0);
  for (int ci = 0; ci < 4; ++ci) {
    if (ci < 3) {
      stage(ci + 1, (ci + 1) & 1);
      asm volatile("s_wait_asynccnt 0x6" ::: "memory");   // chunk ci landed
    } else {
      asm volatile("s_wait_asynccnt 0x0" ::: "memory");
    }
    __syncthreads();

    const __bf16* kc = kch[ci & 1];
    const __bf16* vc = vch[ci & 1];
    for (int m0 = 0; m0 < 256; m0 += 32) {
      Frag ak0, ak1, av0, av1;
      ak0.q[0] = *(const uint4*)(kc + (m0 + li) * 16 + hi * 8);      ak0.q[1] = z4;
      ak1.q[0] = *(const uint4*)(kc + (m0 + 16 + li) * 16 + hi * 8); ak1.q[1] = z4;
      av0.q[0] = *(const uint4*)(vc + li * 256 + m0 + hi * 8);
      av0.q[1] = *(const uint4*)(vc + li * 256 + m0 + 16 + hi * 8);
      av1.q[0] = *(const uint4*)(vc + (16 + li) * 256 + m0 + hi * 8);
      av1.q[1] = *(const uint4*)(vc + (16 + li) * 256 + m0 + 16 + hi * 8);

      step(bq0.v, ak0.v, ak1.v, av0.v, av1.v, O00, O01, mr0, lr0);
      step(bq1.v, ak0.v, ak1.v, av0.v, av1.v, O10, O11, mr1, lr1);
    }
    __syncthreads();   // all waves done reading this buffer
  }

  // epilogue: packed b128 stores into transposed yT[b][n][h*32+c]
  const float i0 = 1.f / lr0, i1 = 1.f / lr1;
  Pack8 pa, pb, pc, pd;
#pragma unroll
  for (int r = 0; r < 8; ++r) {
    pa.e[r] = (__bf16)(O00[r] * i0);
    pb.e[r] = (__bf16)(O01[r] * i0);
    pc.e[r] = (__bf16)(O10[r] * i1);
    pd.e[r] = (__bf16)(O11[r] * i1);
  }
  __bf16* y0 = yT + ((size_t)b * 1024 + tile0 * 16 + li) * 256 + h * 32;
  __bf16* y1 = y0 + 16 * 256;
  *reinterpret_cast<uint4*>(y0 + hi * 8)      = pa.q;
  *reinterpret_cast<uint4*>(y0 + 16 + hi * 8) = pb.q;
  *reinterpret_cast<uint4*>(y1 + hi * 8)      = pc.q;
  *reinterpret_cast<uint4*>(y1 + 16 + hi * 8) = pd.q;
}

// ---------------------------------------------------------------------------
// Depthwise 3x3 conv (SAME) on v + BN, accumulated into yT (bf16 RMW).
// ---------------------------------------------------------------------------
__global__ __launch_bounds__(256)
void pe_kernel(const __bf16* __restrict__ qkvN, const float* __restrict__ wpe,
               const float* __restrict__ gg, const float* __restrict__ bb,
               const float* __restrict__ mm, const float* __restrict__ vv,
               __bf16* __restrict__ yT) {
  const int bc = blockIdx.x;
  const int b = bc >> 8, c = bc & 255;
  const int hg = c >> 5, cv = c & 31;
  const __bf16* vin = qkvN + ((size_t)b * 512 + hg * 64 + 32 + cv) * 1024;
  float w[9];
#pragma unroll
  for (int i = 0; i < 9; ++i) w[i] = wpe[c * 9 + i];
  const float s = gg[c] * rsqrtf(vv[c] + EPSV);
  const float t = bb[c] - mm[c] * s;
  __bf16* yo = yT + (size_t)b * 1024 * 256 + c;

  for (int i = threadIdx.x; i < 1024; i += 256) {
    const int hh = i >> 5, ww = i & 31;
    float acc = 0.f;
#pragma unroll
    for (int kh = -1; kh <= 1; ++kh) {
      const int hy = hh + kh;
      if (hy < 0 || hy > 31) continue;
#pragma unroll
      for (int kw = -1; kw <= 1; ++kw) {
        const int wx = ww + kw;
        if (wx < 0 || wx > 31) continue;
        acc += w[(kh + 1) * 3 + (kw + 1)] * (float)vin[hy * 32 + wx];
      }
    }
    __bf16* p = yo + (size_t)i * 256;
    *p = (__bf16)((float)*p + acc * s + t);
  }
}

// ---------------------------------------------------------------------------
extern "C" void kernel_launch(void* const* d_in, const int* in_sizes, int n_in,
                              void* d_out, int out_size, void* d_ws, size_t ws_size,
                              hipStream_t stream) {
  (void)in_sizes; (void)n_in; (void)out_size; (void)ws_size;
  const float* x      = (const float*)d_in[0];
  const float* w_qkv  = (const float*)d_in[1];
  const float* qkv_g  = (const float*)d_in[2];
  const float* qkv_b  = (const float*)d_in[3];
  const float* qkv_m  = (const float*)d_in[4];
  const float* qkv_v  = (const float*)d_in[5];
  const float* w_pe   = (const float*)d_in[6];
  const float* pe_g   = (const float*)d_in[7];
  const float* pe_b   = (const float*)d_in[8];
  const float* pe_m   = (const float*)d_in[9];
  const float* pe_v   = (const float*)d_in[10];
  const float* w_proj = (const float*)d_in[11];
  const float* proj_g = (const float*)d_in[12];
  const float* proj_b = (const float*)d_in[13];
  const float* proj_m = (const float*)d_in[14];
  const float* proj_v = (const float*)d_in[15];

  char* ws = (char*)d_ws;
  __bf16* XT     = (__bf16*)(ws);                       //  4 MB (B,1024,256)
  __bf16* qkvN   = (__bf16*)(ws + (4u << 20));          //  8 MB (B,512,1024)
  __bf16* qkvT   = (__bf16*)(ws + (12u << 20));         //  8 MB (B,1024,512)
  __bf16* yTbuf  = (__bf16*)(ws + (20u << 20));         //  4 MB (B,1024,256)
  __bf16* wqkvB  = (__bf16*)(ws + (24u << 20));         // 256 KB
  __bf16* wprojB = (__bf16*)(ws + (24u << 20) + (512u << 10));  // 128 KB

  cvtw_kernel<<<dim3(512), 256, 0, stream>>>(w_qkv, wqkvB, 512 * 256);
  cvtw_kernel<<<dim3(256), 256, 0, stream>>>(w_proj, wprojB, 256 * 256);
  xpose_kernel<<<dim3(64, 16, 8), 256, 0, stream>>>(x, XT);

  gemm16_bn_kernel<<<dim3(64, 32, 8), 32, 0, stream>>>(
      XT, wqkvB, qkv_g, qkv_b, qkv_m, qkv_v,
      nullptr, qkvN, qkvT, 256, 512);

  attn_kernel<<<dim3(4, 8, 8), 256, 0, stream>>>(qkvT, qkvN, yTbuf);

  pe_kernel<<<dim3(2048), 256, 0, stream>>>(
      qkvN, w_pe, pe_g, pe_b, pe_m, pe_v, yTbuf);

  gemm16_bn_kernel<<<dim3(64, 16, 8), 32, 0, stream>>>(
      yTbuf, wprojB, proj_g, proj_b, proj_m, proj_v,
      (float*)d_out, nullptr, nullptr, 256, 256);
}
